// Rwkv6SelfAttention_70145405879023
// MI455X (gfx1250) — matrix operand order repaired and checked by reading, not compiled, not run on hardware
//
#include <hip/hip_runtime.h>
#include <hip/hip_bf16.h>

// ---------------------------------------------------------------------------
// RWKV6 self-attention block for MI455X (gfx1250, wave32, WMMA bf16 path)
// ---------------------------------------------------------------------------

typedef __bf16 bhalf;
typedef __attribute__((ext_vector_type(16))) __bf16 v16bf;
typedef __attribute__((ext_vector_type(8)))  float  v8f;

constexpr int B_ = 4, T_ = 2048, C_ = 2048, H_ = 32, N_ = 64;
constexpr int M_ = B_ * T_;                 // 8192 tokens
constexpr float EPS_ = 1e-5f * 64.0f;       // 1e-5 * head_size_divisor**2

__device__ __forceinline__ bhalf f2bf(float f) {
    union { float f; unsigned u; } x; x.f = f;
    unsigned r = (x.u + 0x7FFFu + ((x.u >> 16) & 1u)) >> 16;   // RNE
    unsigned short s = (unsigned short)r;
    bhalf out;
    __builtin_memcpy(&out, &s, 2);
    return out;
}

// CDNA5 async DMA: global -> LDS, 16B per lane, no VGPR staging (ASYNCcnt).
__device__ __forceinline__ void async_ld_b128(unsigned lds, const void* g) {
    asm volatile("global_load_async_to_lds_b128 %0, %1, off" :: "v"(lds), "v"(g) : "memory");
}
__device__ __forceinline__ void async_ld_b128_o16(unsigned lds, const void* g) {
    asm volatile("global_load_async_to_lds_b128 %0, %1, off offset:16" :: "v"(lds), "v"(g) : "memory");
}
__device__ __forceinline__ void wait_async0() {
    asm volatile("s_wait_asynccnt 0x0" ::: "memory");
}

// ---------------------------------------------------------------------------
// 0) f32 -> bf16 weight cast
// ---------------------------------------------------------------------------
__global__ void k_cast(const float* __restrict__ src, bhalf* __restrict__ dst, size_t n) {
    for (size_t i = (size_t)blockIdx.x * blockDim.x + threadIdx.x; i < n;
         i += (size_t)gridDim.x * blockDim.x)
        dst[i] = f2bf(src[i]);
}

// ---------------------------------------------------------------------------
// 1) token shift: xx = shift(x) - x ; xm = x + xx * time_maa_x
// ---------------------------------------------------------------------------
__global__ void k_prep(const float* __restrict__ x, const float* __restrict__ maa_x,
                       float* __restrict__ xx, float* __restrict__ xm, size_t total) {
    for (size_t i = (size_t)blockIdx.x * blockDim.x + threadIdx.x; i < total;
         i += (size_t)gridDim.x * blockDim.x) {
        size_t c  = i % C_;
        size_t bt = i / C_;
        size_t t  = bt % T_;
        float cur  = x[i];
        float prev = (t == 0) ? 0.f : x[i - C_];
        float d = prev - cur;
        xx[i] = d;
        xm[i] = fmaf(d, maa_x[c], cur);
    }
}

// ---------------------------------------------------------------------------
// 2) t5 = tanh(xm @ W1)   [M,2048] @ [2048,160]
// ---------------------------------------------------------------------------
__global__ __launch_bounds__(160)
void k_lr1(const float* __restrict__ xm, const float* __restrict__ w1,
           float* __restrict__ t5) {
    const int row = blockIdx.x;
    const int col = threadIdx.x;                 // 0..159
    const float* xr = xm + (size_t)row * C_;
    float acc = 0.f;
    #pragma unroll 8
    for (int k = 0; k < C_; ++k) acc = fmaf(xr[k], w1[k * 160 + col], acc);
    t5[(size_t)row * 160 + col] = tanhf(acc);
}

// ---------------------------------------------------------------------------
// 3) m_f = t5 @ W2[f] ; x? = x + xx*(maa_? + m_?)  -> xd(f32), xk/xv/xr/xg(bf16)
// ---------------------------------------------------------------------------
__global__ __launch_bounds__(256)
void k_mix(const float* __restrict__ x, const float* __restrict__ xx,
           const float* __restrict__ t5, const float* __restrict__ w2,
           const float* __restrict__ mw, const float* __restrict__ mk,
           const float* __restrict__ mv, const float* __restrict__ mr,
           const float* __restrict__ mg,
           float* __restrict__ xd, bhalf* __restrict__ xk, bhalf* __restrict__ xv,
           bhalf* __restrict__ xr, bhalf* __restrict__ xg) {
    const int row = blockIdx.x;
    const int tid = threadIdx.x;
    __shared__ float ts[160];
    if (tid < 160) ts[tid] = t5[(size_t)row * 160 + tid];
    __syncthreads();

    float xv0[8], xxv[8];
    #pragma unroll
    for (int j = 0; j < 8; ++j) {
        size_t idx = (size_t)row * C_ + tid + j * 256;
        xv0[j] = x[idx]; xxv[j] = xx[idx];
    }
    const float* maas[5] = {mw, mk, mv, mr, mg};
    for (int f = 0; f < 5; ++f) {
        float acc[8];
        #pragma unroll
        for (int j = 0; j < 8; ++j) acc[j] = 0.f;
        #pragma unroll 4
        for (int e = 0; e < 32; ++e) {
            float tv = ts[f * 32 + e];
            const float* wrow = w2 + (size_t)(f * 32 + e) * C_;
            #pragma unroll
            for (int j = 0; j < 8; ++j)
                acc[j] = fmaf(tv, wrow[tid + j * 256], acc[j]);
        }
        #pragma unroll
        for (int j = 0; j < 8; ++j) {
            int c = tid + j * 256;
            size_t idx = (size_t)row * C_ + c;
            float o = fmaf(xxv[j], maas[f][c] + acc[j], xv0[j]);
            if      (f == 0) xd[idx] = o;
            else if (f == 1) xk[idx] = f2bf(o);
            else if (f == 2) xv[idx] = f2bf(o);
            else if (f == 3) xr[idx] = f2bf(o);
            else             xg[idx] = f2bf(o);
        }
    }
}

// ---------------------------------------------------------------------------
// 4) w = exp(-exp(time_decay + tanh(xd @ dw1) @ dw2))
// ---------------------------------------------------------------------------
__global__ __launch_bounds__(256)
void k_decay(const float* __restrict__ xd, const float* __restrict__ td0,
             const float* __restrict__ dw1, const float* __restrict__ dw2,
             float* __restrict__ wdec) {
    const int row = blockIdx.x;
    const int tid = threadIdx.x;
    const int e = tid & 63, kg = tid >> 6;
    __shared__ float part[256];
    __shared__ float hs[64];
    const float* xr = xd + (size_t)row * C_;
    float p = 0.f;
    #pragma unroll 8
    for (int k = kg * 512; k < (kg + 1) * 512; ++k)
        p = fmaf(xr[k], dw1[k * 64 + e], p);
    part[tid] = p;
    __syncthreads();
    if (tid < 64)
        hs[tid] = tanhf(part[tid] + part[tid + 64] + part[tid + 128] + part[tid + 192]);
    __syncthreads();
    #pragma unroll
    for (int j = 0; j < 8; ++j) {
        int c = tid + j * 256;
        float acc = td0[c];
        #pragma unroll 8
        for (int ee = 0; ee < 64; ++ee)
            acc = fmaf(hs[ee], dw2[ee * C_ + c], acc);
        wdec[(size_t)row * C_ + c] = __expf(-__expf(acc));
    }
}

// ---------------------------------------------------------------------------
// 5) WMMA bf16 NT GEMM (fixed 8192 x 2048 x 2048):
//    C[M,N] (f32) = A[M,K] (bf16) x B[N,K]^T (bf16)
//    128x128 block tile, BK=32, 8 waves, 2x4 16x16 accum tiles per wave.
//    Tiles staged with global_load_async_to_lds_b128 (ASYNCcnt), double buffer.
// ---------------------------------------------------------------------------
#define BM 128
#define BN 128
#define BK 32
#define LDT 40                        // padded LDS row stride (halves)
#define BUFB (BM * LDT * 2)           // bytes per LDS buffer (10240)

union Frag { v16bf v; uint4 q[2]; };

__global__ __launch_bounds__(256)
void k_gemm(const bhalf* __restrict__ A, const bhalf* __restrict__ Bm,
            float* __restrict__ Cout) {
    constexpr int K = C_;             // 2048
    constexpr int N = C_;             // 2048
    constexpr int KT = K / BK;        // 64

    __shared__ bhalf As[2][BM * LDT];
    __shared__ bhalf Bs[2][BN * LDT];

    const int tid  = threadIdx.x;
    const int lane = tid & 31;
    const int wave = tid >> 5;
    const int waveM = (wave >> 1) * 32;      // 4 wave-rows
    const int waveN = (wave & 1) * 64;       // 2 wave-cols
    const int bM = blockIdx.x * BM;
    const int bN = blockIdx.y * BN;

    // staging map: each thread owns 32B (16 bf16) of one row; two async b128s
    const int lrow = tid >> 1;               // 0..127
    const int lk   = (tid & 1) * 16;         // 0 or 16
    const bhalf* aptr = A  + (size_t)(bM + lrow) * K + lk;
    const bhalf* bptr = Bm + (size_t)(bN + lrow) * K + lk;
    const unsigned ldsA0 = (unsigned)(uintptr_t)&As[0][lrow * LDT + lk];
    const unsigned ldsB0 = (unsigned)(uintptr_t)&Bs[0][lrow * LDT + lk];

    // prologue: DMA tile 0 into buffer 0
    async_ld_b128    (ldsA0, aptr);
    async_ld_b128_o16(ldsA0, aptr);
    async_ld_b128    (ldsB0, bptr);
    async_ld_b128_o16(ldsB0, bptr);
    wait_async0();
    __syncthreads();

    v8f acc[2][4];
    const v8f vzero = {0.f, 0.f, 0.f, 0.f, 0.f, 0.f, 0.f, 0.f};
    #pragma unroll
    for (int i = 0; i < 2; ++i)
        #pragma unroll
        for (int j = 0; j < 4; ++j) acc[i][j] = vzero;

    const int arow = lane & 15;
    const int ah   = lane >> 4;              // which half-wave

    for (int kt = 0; kt < KT; ++kt) {
        const int cur = kt & 1;
        // kick DMA for next tile into the other buffer, overlapped with WMMA
        if (kt + 1 < KT) {
            const unsigned nb = (unsigned)((kt + 1) & 1) * BUFB;
            const bhalf* ap = aptr + (kt + 1) * BK;
            const bhalf* bp = bptr + (kt + 1) * BK;
            async_ld_b128    (ldsA0 + nb, ap);
            async_ld_b128_o16(ldsA0 + nb, ap);
            async_ld_b128    (ldsB0 + nb, bp);
            async_ld_b128_o16(ldsB0 + nb, bp);
        }
        // A fragments: lane holds row m, K chunks {ah*8..+7, 16+ah*8..+7}
        Frag af[2], bf[4];
        #pragma unroll
        for (int i = 0; i < 2; ++i) {
            const bhalf* p = &As[cur][(waveM + i * 16 + arow) * LDT + ah * 8];
            af[i].q[0] = *(const uint4*)p;
            af[i].q[1] = *(const uint4*)(p + 16);
        }
        // B fragments: lane holds col n, K chunks {ah*8..+7, 16+ah*8..+7}
        #pragma unroll
        for (int j = 0; j < 4; ++j) {
            const bhalf* p = &Bs[cur][(waveN + j * 16 + arow) * LDT + ah * 8];
            bf[j].q[0] = *(const uint4*)p;
            bf[j].q[1] = *(const uint4*)(p + 16);
        }
        #pragma unroll
        for (int i = 0; i < 2; ++i)
            #pragma unroll
            for (int j = 0; j < 4; ++j)
                acc[i][j] = __builtin_amdgcn_wmma_f32_16x16x32_bf16(
                    false, af[i].v, false, bf[j].v, (short)0, acc[i][j], false, false);

        if (kt + 1 < KT) wait_async0();
        __syncthreads();
    }

    // C/D layout: VGPR r -> M = r (lanes 0-15) or r+8 (lanes 16-31); N = lane&15
    // Fixed N stride => single base register + immediate offsets for all stores.
    const int mhalf = (lane < 16) ? 0 : 8;
    const int ccol  = lane & 15;
    float* cbase = Cout + (size_t)(bM + waveM + mhalf) * N + (bN + waveN + ccol);
    #pragma unroll
    for (int i = 0; i < 2; ++i)
        #pragma unroll
        for (int j = 0; j < 4; ++j) {
            float* p = cbase + i * 16 * N + j * 16;
            #pragma unroll
            for (int r = 0; r < 8; ++r)
                p[r * N] = acc[i][j][r];
        }
}

// ---------------------------------------------------------------------------
// 6) WKV scan: one block per (b,h); 64x64 state in registers (16 rows/thread)
// ---------------------------------------------------------------------------
__global__ __launch_bounds__(256)
void k_wkv(const float* __restrict__ r, const float* __restrict__ k,
           const float* __restrict__ v, const float* __restrict__ wdec,
           const float* __restrict__ u, float* __restrict__ att) {
    const int bh = blockIdx.x;
    const int b = bh / H_, h = bh % H_;
    const int tid = threadIdx.x;
    const int m  = tid & 63;     // value index owned by this thread
    const int g  = tid >> 6;     // key-group 0..3
    const int nb = g * 16;       // first key row owned

    __shared__ float rs[64], ks[64], vs[64], wsv[64], red[256], pruk[4];

    float st[16];
    #pragma unroll
    for (int i = 0; i < 16; ++i) st[i] = 0.f;
    float uu[16];
    #pragma unroll
    for (int i = 0; i < 16; ++i) uu[i] = u[h * 64 + nb + i];

    for (int t = 0; t < T_; ++t) {
        const size_t base = ((size_t)(b * T_ + t)) * C_ + h * 64;
        if      (tid < 64)  rs[tid]        = r[base + tid];
        else if (tid < 128) ks[tid - 64]   = k[base + tid - 64];
        else if (tid < 192) vs[tid - 128]  = v[base + tid - 128];
        else                wsv[tid - 192] = wdec[base + tid - 192];
        __syncthreads();

        const float vm = vs[m];
        float p = 0.f, pu = 0.f;
        #pragma unroll
        for (int i = 0; i < 16; ++i) {
            int n = nb + i;
            float rn = rs[n], kn = ks[n], wn = wsv[n];
            float s = st[i];
            p  = fmaf(rn, s, p);
            pu = fmaf(rn * uu[i], kn, pu);
            st[i] = fmaf(kn, vm, wn * s);
        }
        red[tid] = p;
        if (m == 0) pruk[g] = pu;
        __syncthreads();
        if (tid < 64) {
            float ruk = pruk[0] + pruk[1] + pruk[2] + pruk[3];
            float o = red[tid] + red[tid + 64] + red[tid + 128] + red[tid + 192];
            att[base + tid] = fmaf(vs[tid], ruk, o);
        }
        __syncthreads();
    }
}

// ---------------------------------------------------------------------------
// 7) GroupNorm (per b,t,h over N=64) * silu(g) -> bf16
// ---------------------------------------------------------------------------
__global__ __launch_bounds__(256)
void k_gnorm(const float* __restrict__ att, const float* __restrict__ gsrc,
             const float* __restrict__ lnw, const float* __restrict__ lnb,
             bhalf* __restrict__ y) {
    const int tid = threadIdx.x;
    const int seg = tid >> 6;              // head within block
    const int l   = tid & 63;
    const int hid = blockIdx.x * 4 + seg;  // global (token,head)
    const int token = hid >> 5;            // / H_
    const int h     = hid & 31;
    const size_t idx = (size_t)token * C_ + h * 64 + l;

    float val = att[idx];
    __shared__ float s1[256], s2[256];
    s1[tid] = val; s2[tid] = val * val;
    __syncthreads();
    #pragma unroll
    for (int off = 32; off >= 1; off >>= 1) {
        if (l < off) { s1[tid] += s1[tid + off]; s2[tid] += s2[tid + off]; }
        __syncthreads();
    }
    float mu  = s1[seg * 64] * (1.f / 64.f);
    float var = s2[seg * 64] * (1.f / 64.f) - mu * mu;
    float inv = rsqrtf(var + EPS_);
    float yv  = (val - mu) * inv * lnw[h * 64 + l] + lnb[h * 64 + l];
    float gv  = gsrc[idx];
    float sil = gv / (1.f + __expf(-gv));
    y[idx] = f2bf(yv * sil);
}

// ---------------------------------------------------------------------------
// Host launcher
// ---------------------------------------------------------------------------
extern "C" void kernel_launch(void* const* d_in, const int* in_sizes, int n_in,
                              void* d_out, int out_size, void* d_ws, size_t ws_size,
                              hipStream_t stream) {
    (void)in_sizes; (void)n_in; (void)out_size; (void)ws_size;
    const float* x     = (const float*)d_in[0];
    const float* maa_x = (const float*)d_in[1];
    const float* maa_w = (const float*)d_in[2];
    const float* maa_k = (const float*)d_in[3];
    const float* maa_v = (const float*)d_in[4];
    const float* maa_r = (const float*)d_in[5];
    const float* maa_g = (const float*)d_in[6];
    const float* w1    = (const float*)d_in[7];
    const float* w2    = (const float*)d_in[8];
    const float* tdec  = (const float*)d_in[9];
    const float* dw1   = (const float*)d_in[10];
    const float* dw2   = (const float*)d_in[11];
    const float* u     = (const float*)d_in[12];
    const float* w_r   = (const float*)d_in[13];
    const float* w_k   = (const float*)d_in[14];
    const float* w_v   = (const float*)d_in[15];
    const float* w_g   = (const float*)d_in[16];
    const float* w_o   = (const float*)d_in[17];
    const float* lnw   = (const float*)d_in[18];
    const float* lnb   = (const float*)d_in[19];

    char* ws = (char*)d_ws;
    size_t o = 0;
    auto take = [&](size_t bytes) -> void* {
        void* p = ws + o;
        o += (bytes + 255) & ~size_t(255);
        return p;
    };
    const size_t fMC = (size_t)M_ * C_ * sizeof(float);
    const size_t hMC = (size_t)M_ * C_ * sizeof(bhalf);
    const size_t wsz = (size_t)C_ * C_;

    float* xx   = (float*)take(fMC);
    float* xm   = (float*)take(fMC);
    float* t5   = (float*)take((size_t)M_ * 160 * sizeof(float));
    float* xd   = (float*)take(fMC);
    bhalf* xkb  = (bhalf*)take(hMC);
    bhalf* xvb  = (bhalf*)take(hMC);
    bhalf* xrb  = (bhalf*)take(hMC);
    bhalf* xgb  = (bhalf*)take(hMC);
    float* wdec = (float*)take(fMC);
    float* kb   = (float*)take(fMC);
    float* gb   = (float*)take(fMC);
    bhalf* y    = (bhalf*)take(hMC);
    bhalf* wrb  = (bhalf*)take(wsz * sizeof(bhalf));
    bhalf* wkb  = (bhalf*)take(wsz * sizeof(bhalf));
    bhalf* wvb  = (bhalf*)take(wsz * sizeof(bhalf));
    bhalf* wgb  = (bhalf*)take(wsz * sizeof(bhalf));
    bhalf* wob  = (bhalf*)take(wsz * sizeof(bhalf));
    // aliases (lifetimes do not overlap):
    float* rb   = xx;   // xx dead after k_mix; r produced after
    float* att  = xm;   // xm dead after k_lr1
    float* vb   = xd;   // xd dead after k_decay

    // 0) weight casts
    k_cast<<<2048, 256, 0, stream>>>(w_r, wrb, wsz);
    k_cast<<<2048, 256, 0, stream>>>(w_k, wkb, wsz);
    k_cast<<<2048, 256, 0, stream>>>(w_v, wvb, wsz);
    k_cast<<<2048, 256, 0, stream>>>(w_g, wgb, wsz);
    k_cast<<<2048, 256, 0, stream>>>(w_o, wob, wsz);

    // 1) token shift + maa_x mix
    k_prep<<<65536, 256, 0, stream>>>(x, maa_x, xx, xm, (size_t)M_ * C_);
    // 2) low-rank lerp stage 1
    k_lr1<<<M_, 160, 0, stream>>>(xm, w1, t5);
    // 3) low-rank lerp stage 2 + build xd/xk/xv/xr/xg
    k_mix<<<M_, 256, 0, stream>>>(x, xx, t5, w2, maa_w, maa_k, maa_v, maa_r, maa_g,
                                  xd, xkb, xvb, xrb, xgb);
    // 4) per-channel decay
    k_decay<<<M_, 256, 0, stream>>>(xd, tdec, dw1, dw2, wdec);

    // 5) four big WMMA projections
    dim3 gg(M_ / BM, C_ / BN);
    k_gemm<<<gg, 256, 0, stream>>>(xrb, wrb, rb);
    k_gemm<<<gg, 256, 0, stream>>>(xkb, wkb, kb);
    k_gemm<<<gg, 256, 0, stream>>>(xvb, wvb, vb);
    k_gemm<<<gg, 256, 0, stream>>>(xgb, wgb, gb);

    // 6) WKV recurrence
    k_wkv<<<B_ * H_, 256, 0, stream>>>(rb, kb, vb, wdec, u, att);

    // 7) GroupNorm * silu(g) -> bf16
    k_gnorm<<<(M_ * H_) / 4, 256, 0, stream>>>(att, gb, lnw, lnb, y);

    // 8) output projection (f32 result straight to d_out)
    k_gemm<<<gg, 256, 0, stream>>>(y, wob, (float*)d_out);
}
